// GETModel_80857054315097
// MI455X (gfx1250) — matrix operand order repaired and checked by reading, not compiled, not run on hardware
//
#include <hip/hip_runtime.h>
#include <hip/hip_bf16.h>
#include <math.h>

// ---------------- CDNA5 WMMA types ----------------
typedef __attribute__((ext_vector_type(16))) _Float16 v16h;
typedef __attribute__((ext_vector_type(8)))  float    v8f;
typedef __attribute__((ext_vector_type(4)))  float    f4;

__device__ inline float waveReduceSum(float v) {
#pragma unroll
  for (int o = 16; o > 0; o >>= 1) v += __shfl_xor(v, o, 32);
  return v;
}
__device__ inline float waveReduceMax(float v) {
#pragma unroll
  for (int o = 16; o > 0; o >>= 1) v = fmaxf(v, __shfl_xor(v, o, 32));
  return v;
}
__device__ inline float softplusf(float x) { return (x > 20.f) ? x : log1pf(expf(x)); }
__device__ inline float sigmoidf_(float x) { return 1.f / (1.f + expf(-x)); }
__device__ inline float geluf(float x) { return 0.5f * x * (1.f + erff(x * 0.70710678118654752f)); }
__device__ inline void atomAdd(float* p, float v) {
  __hip_atomic_fetch_add(p, v, __ATOMIC_RELAXED, __HIP_MEMORY_SCOPE_AGENT);
}

// 4x b128 loads of one 32-element K-chunk
__device__ inline void load_chunkA(const float* p, int aKb, f4* r) {
  r[0] = *(const f4*)(p + aKb);
  r[1] = *(const f4*)(p + aKb + 4);
  r[2] = *(const f4*)(p + 8 + aKb);
  r[3] = *(const f4*)(p + 12 + aKb);
}
__device__ inline void load_chunkB(const float* p, f4* r) {
  r[0] = *(const f4*)(p);
  r[1] = *(const f4*)(p + 4);
  r[2] = *(const f4*)(p + 8);
  r[3] = *(const f4*)(p + 12);
}
__device__ inline v16h cvt_chunk(const f4* r) {
  v16h h;
#pragma unroll
  for (int j = 0; j < 4; ++j)
#pragma unroll
    for (int e = 0; e < 4; ++e) h[j * 4 + e] = (_Float16)r[j][e];
  return h;
}

// ---------------- generic WMMA GEMM: C[M,Nc] = act(A[M,K] @ Bt^T + bias) ----------------
// Bt is B transposed ([Nc,K]): both fragments load as contiguous 128-bit vectors.
// One wave computes a 16 x (16*NT) strip. Ping-pong double buffer (manual unroll
// by two 32-K chunks, K must be a multiple of 64): buffer sets are distinct named
// registers and every load is unconditional (prefetch offset clamped on the last
// iteration), so the allocator needs no cross-iteration merge copies.
template <int NT, int ACT, bool HAS_BIAS, bool ACCUM>
__global__ __launch_bounds__(256)
void gemm_wmma_kernel(const float* __restrict__ A, const float* __restrict__ Bt,
                      const float* __restrict__ bias, float* __restrict__ C,
                      int M, int Nc, int K) {
  int wave = (blockIdx.x * blockDim.x + threadIdx.x) >> 5;
  int lane = threadIdx.x & 31;
  int ntm = (M + 15) >> 4;
  int ntn = (Nc + 15) >> 4;
  int ntg = (ntn + NT - 1) / NT;
  if (wave >= ntm * ntg) return;
  int tm = wave / ntg;
  int tnb = (wave % ntg) * NT;

  int mIdx = (lane & 15);
  int grp  = lane >> 4;                 // half-wave select
  int aKb  = grp * 16;                  // A frag: e -> K = aKb + e (contiguous)
  int bKb  = grp * 16;                  // B frag: e -> K = bKb + e (contiguous in Bt)

  int row = tm * 16 + mIdx;
  if (row >= M) row = M - 1;
  const float* arow0 = A + (size_t)row * K;

  const float* brow[NT];
#pragma unroll
  for (int nt = 0; nt < NT; ++nt) {
    int col = (tnb + nt) * 16 + mIdx;
    if (col >= Nc) col = Nc - 1;        // clamp: col N only feeds output col N (guarded)
    brow[nt] = Bt + (size_t)col * K + bKb;
  }

  v8f acc[NT];
#pragma unroll
  for (int nt = 0; nt < NT; ++nt) acc[nt] = (v8f){0.f,0.f,0.f,0.f,0.f,0.f,0.f,0.f};

  f4 ar0[4], br0[NT][4];   // buffer set 0 (even chunks)
  f4 ar1[4], br1[NT][4];   // buffer set 1 (odd chunks)

  load_chunkA(arow0, aKb, ar0);
#pragma unroll
  for (int nt = 0; nt < NT; ++nt) load_chunkB(brow[nt], br0[nt]);

  for (int kk = 0; kk < K; kk += 64) {
    // odd chunk load is always in-range (K % 64 == 0, kk <= K-64)
    int k1 = kk + 32;
    load_chunkA(arow0 + k1, aKb, ar1);
#pragma unroll
    for (int nt = 0; nt < NT; ++nt) load_chunkB(brow[nt] + k1, br1[nt]);

    {  // consume even chunk
      v16h af = cvt_chunk(ar0);
#pragma unroll
      for (int nt = 0; nt < NT; ++nt) {
        v16h bf = cvt_chunk(br0[nt]);
        acc[nt] = __builtin_amdgcn_wmma_f32_16x16x32_f16(false, af, false, bf, (short)0,
                                                         acc[nt], false, false);
      }
    }

    // prefetch next even chunk; clamp offset on the final iteration (unused data)
    int k2 = kk + 64;
    if (k2 >= K) k2 = 0;
    load_chunkA(arow0 + k2, aKb, ar0);
#pragma unroll
    for (int nt = 0; nt < NT; ++nt) load_chunkB(brow[nt] + k2, br0[nt]);

    {  // consume odd chunk
      v16h af = cvt_chunk(ar1);
#pragma unroll
      for (int nt = 0; nt < NT; ++nt) {
        v16h bf = cvt_chunk(br1[nt]);
        acc[nt] = __builtin_amdgcn_wmma_f32_16x16x32_f16(false, af, false, bf, (short)0,
                                                         acc[nt], false, false);
      }
    }
  }

#pragma unroll
  for (int nt = 0; nt < NT; ++nt) {
#pragma unroll
    for (int r = 0; r < 8; ++r) {
      int orow = tm * 16 + r + grp * 8;
      int ocol = (tnb + nt) * 16 + mIdx;
      if (orow < M && ocol < Nc) {
        float v = acc[nt][r];
        if (HAS_BIAS) v += bias[ocol];
        if (ACT == 1) v = geluf(v);
        size_t idx = (size_t)orow * Nc + ocol;
        if (ACCUM) C[idx] += v; else C[idx] = v;
      }
    }
  }
}

// ---------------- row LayerNorm (in-place safe), one wave per row, W in {128,256} --------
__global__ void ln_rows_kernel(const float* __restrict__ src, float* __restrict__ dst,
                               const float* __restrict__ g, const float* __restrict__ b,
                               int Nrows, int W, float* __restrict__ mean_out,
                               float* __restrict__ rstd_out) {
  int wave = (blockIdx.x * blockDim.x + threadIdx.x) >> 5;
  int lane = threadIdx.x & 31;
  if (wave >= Nrows) return;
  const float* x = src + (size_t)wave * W;
  int epl = W >> 5;  // 4 or 8
  float vals[8];
  float s = 0.f;
  for (int i = 0; i < epl; ++i) { vals[i] = x[lane + 32 * i]; s += vals[i]; }
  s = waveReduceSum(s);
  float m = s / (float)W;
  float var = 0.f;
  for (int i = 0; i < epl; ++i) { float d = vals[i] - m; var += d * d; }
  var = waveReduceSum(var) / (float)W;
  float rstd = rsqrtf(var + 1e-5f);
  float* y = dst + (size_t)wave * W;
  for (int i = 0; i < epl; ++i) {
    int c = lane + 32 * i;
    y[c] = (vals[i] - m) * rstd * g[c] + b[c];
  }
  if (mean_out && lane == 0) { mean_out[wave] = m; rstd_out[wave] = rstd; }
}

// ---------------- weight packing -------------------------------------------------
// Wf[128,640]: Wf[k][c*128+d] = Wc[k][d]   (Bt for the BACKWARD GEMM)
// Wb[640,128]: Wb[c*128+d][k] = Wc[k][d]   (Bt for the FORWARD  GEMM)
__global__ void pack_weights_kernel(const float* __restrict__ Wq2, const float* __restrict__ Wk2,
                                    const float* __restrict__ Wq3, const float* __restrict__ Wk3,
                                    const float* __restrict__ Wqm, float* __restrict__ Wf,
                                    float* __restrict__ Wb) {
  int idx = blockIdx.x * blockDim.x + threadIdx.x;
  if (idx >= 5 * 128 * 128) return;
  int c = idx / (128 * 128);
  int rem = idx % (128 * 128);
  int k = rem / 128, d = rem % 128;
  const float* W = (c == 0) ? Wq2 : (c == 1) ? Wk2 : (c == 2) ? Wq3 : (c == 3) ? Wk3 : Wqm;
  float v = W[k * 128 + d];
  Wf[(size_t)k * 640 + c * 128 + d] = v;
  Wb[(size_t)(c * 128 + d) * 128 + k] = v;
}

// ---------------- W[R,C] -> Wt[C,R] ----------------
__global__ void transpose_kernel(const float* __restrict__ W, float* __restrict__ Wt,
                                 int R, int C) {
  int idx = blockIdx.x * blockDim.x + threadIdx.x;
  if (idx >= R * C) return;
  int r = idx / C, c = idx % C;
  Wt[(size_t)c * R + r] = W[idx];
}

// ---------------- Km = B_mem[32,128] @ W_Km[128,128] ----------------
__global__ void km_kernel(const float* __restrict__ Bmem, const float* __restrict__ Wkm,
                          float* __restrict__ Km) {
  int d = threadIdx.x;   // 128
  int r = blockIdx.x;    // 32
  float acc = 0.f;
  for (int k = 0; k < 128; ++k) acc += Bmem[r * 128 + k] * Wkm[k * 128 + d];
  Km[r * 128 + d] = acc;
}

// ---------------- a_2[e] = GELU(edge_attr@em_W1+b1) @ em_W2 + b2 ----------------
__global__ void edge_bias_kernel(const float* __restrict__ eattr, const float* __restrict__ W1,
                                 const float* __restrict__ b1, const float* __restrict__ W2,
                                 const float* __restrict__ b2, float* __restrict__ a2, int E) {
  int wave = (blockIdx.x * blockDim.x + threadIdx.x) >> 5;
  int lane = threadIdx.x & 31;
  if (wave >= E) return;
  const float* ea = eattr + (size_t)wave * 5;
  float e0 = ea[0], e1 = ea[1], e2 = ea[2], e3 = ea[3], e4 = ea[4];
  float part = 0.f;
#pragma unroll
  for (int i = 0; i < 4; ++i) {
    int d = lane + 32 * i;
    float h = e0 * W1[d] + e1 * W1[128 + d] + e2 * W1[256 + d] + e3 * W1[384 + d] +
              e4 * W1[512 + d] + b1[d];
    part += geluf(h) * W2[d];
  }
  part = waveReduceSum(part);
  if (lane == 0) a2[wave] = part + b2[0];
}

// ---------------- 2-body edge energy gradient (one wave per edge) ----------------
// QK layout per row (stride 640): [Q2 | K2 | Q3 | K3 | Qm]
__global__ void edge2_kernel(const int* __restrict__ c2, const float* __restrict__ u2,
                             const float* __restrict__ a2, const float* __restrict__ QK,
                             float* __restrict__ dQK, const float* __restrict__ l2p,
                             const float* __restrict__ b2p, int E) {
  int wave = (blockIdx.x * blockDim.x + threadIdx.x) >> 5;
  int lane = threadIdx.x & 31;
  if (wave >= E) return;
  int s = c2[wave], d = c2[E + wave];
  const float* q = QK + (size_t)s * 640;
  const float* k = QK + (size_t)d * 640 + 128;
  float qv[4], kv[4], acc = 0.f;
#pragma unroll
  for (int i = 0; i < 4; ++i) {
    qv[i] = q[lane + 32 * i];
    kv[i] = k[lane + 32 * i];
    acc += qv[i] * kv[i];
  }
  acc = waveReduceSum(acc);
  float lam2 = softplusf(l2p[0]);
  float b2 = fminf(softplusf(b2p[0]), 5.f);
  const float isd = 0.088388347648318447f;  // 1/sqrt(128)
  float s2 = b2 * acc * isd + a2[wave];
  float w = -(lam2 * u2[wave] * sigmoidf_(s2)) * isd;  // dE/dQ2 = w*K2 ; dE/dK2 = w*Q2
  float* dq = dQK + (size_t)s * 640;
  float* dk = dQK + (size_t)d * 640 + 128;
#pragma unroll
  for (int i = 0; i < 4; ++i) {
    atomAdd(dq + lane + 32 * i, w * kv[i]);
    atomAdd(dk + lane + 32 * i, w * qv[i]);
  }
}

// ---------------- 3-body edge energy gradient (one wave per edge, R=1) ----------------
__global__ void edge3_kernel(const int* __restrict__ c3, const int* __restrict__ v3,
                             const float* __restrict__ u3, const int* __restrict__ tt,
                             const float* __restrict__ Ttau, const float* __restrict__ G,
                             const float* __restrict__ QK, float* __restrict__ dQK,
                             float* __restrict__ dG, const float* __restrict__ l3p,
                             const float* __restrict__ b3p, int E) {
  int wave = (blockIdx.x * blockDim.x + threadIdx.x) >> 5;
  int lane = threadIdx.x & 31;
  if (wave >= E) return;
  int i0 = c3[wave], j = c3[E + wave], v = v3[wave], t = tt[wave];
  const float* q = QK + (size_t)i0 * 640 + 256;  // Q3
  const float* k = QK + (size_t)j * 640 + 384;   // K3
  const float* g = G + (size_t)v * 128;
  const float* T = Ttau + (size_t)t * 128;
  float qv[4], kv[4], tv[4];
  float qk = 0.f, tg = 0.f;
#pragma unroll
  for (int i = 0; i < 4; ++i) {
    int c = lane + 32 * i;
    qv[i] = q[c]; kv[i] = k[c]; tv[i] = T[c];
    qk += qv[i] * kv[i];
    tg += tv[i] * g[c];
  }
  qk = waveReduceSum(qk);
  tg = waveReduceSum(tg);
  float lam3 = softplusf(l3p[0]);
  float b3 = fminf(softplusf(b3p[0]), 5.f);
  const float invD = 0.0078125f;  // 1/128
  float s3 = b3 * qk * tg * invD;
  float cb = -(lam3 * u3[wave] * sigmoidf_(s3)) * invD;
  float dqk = cb * tg, dtg = cb * qk;
  float* dq = dQK + (size_t)i0 * 640 + 256;
  float* dk = dQK + (size_t)j * 640 + 384;
  float* dg = dG + (size_t)v * 128;
#pragma unroll
  for (int i = 0; i < 4; ++i) {
    int c = lane + 32 * i;
    atomAdd(dq + c, dqk * kv[i]);
    atomAdd(dk + c, dqk * qv[i]);
    atomAdd(dg + c, dtg * tv[i]);
  }
}

// ---------------- memory attention gradient: dQm = -(lamm/sqrt_d) * softmax(sm) @ Km ----
__global__ void mem_kernel(const float* __restrict__ QK, const float* __restrict__ Km,
                           float* __restrict__ dQK, const float* __restrict__ lmp,
                           const float* __restrict__ bmp, int Nrows) {
  __shared__ float sKm[32 * 129];  // padded stride -> conflict-free LDS
  __shared__ float sP[8][32];
  for (int i = threadIdx.x; i < 32 * 128; i += blockDim.x)
    sKm[(i >> 7) * 129 + (i & 127)] = Km[i];
  __syncthreads();
  int wave = (blockIdx.x * blockDim.x + threadIdx.x) >> 5;
  int w = (threadIdx.x >> 5) & 7;
  int lane = threadIdx.x & 31;
  if (wave >= Nrows) return;
  float lamm = softplusf(lmp[0]);
  float bm = fminf(softplusf(bmp[0]), 5.f);
  const float isd = 0.088388347648318447f;
  const float* q = QK + (size_t)wave * 640 + 512;
  float dotv = 0.f;
  for (int d = 0; d < 128; ++d) dotv += q[d] * sKm[lane * 129 + d];
  float sm = bm * dotv * isd;
  float mx = waveReduceMax(sm);
  float e = expf(sm - mx);
  float ssum = waveReduceSum(e);
  sP[w][lane] = e / ssum;  // same-wave LDS ops are in-order
  float* dq = dQK + (size_t)wave * 640 + 512;
#pragma unroll
  for (int j = 0; j < 4; ++j) {
    int d = lane + 32 * j;
    float accv = 0.f;
    for (int k = 0; k < 32; ++k) accv += sP[w][k] * sKm[k * 129 + d];
    dq[d] = -(lamm * isd) * accv;
  }
}

// ---------------- LN backward + clip + descent step (one wave per row) ----------------
__global__ void grad_step_kernel(float* __restrict__ X, const float* __restrict__ dG,
                                 const float* __restrict__ meanv, const float* __restrict__ rstdv,
                                 const float* __restrict__ ln_g, const float* __restrict__ eta_l,
                                 int Nrows) {
  int wave = (blockIdx.x * blockDim.x + threadIdx.x) >> 5;
  int lane = threadIdx.x & 31;
  if (wave >= Nrows) return;
  float eta = 0.25f * sigmoidf_(eta_l[0]) * 0.9999f;
  float m = meanv[wave], rstd = rstdv[wave];
  const float* xr = X + (size_t)wave * 128;
  const float* gr = dG + (size_t)wave * 128;
  float xh[4], dxh[4], xv[4];
  float s1 = 0.f, s2 = 0.f;
#pragma unroll
  for (int i = 0; i < 4; ++i) {
    int c = lane + 32 * i;
    xv[i] = xr[c];
    xh[i] = (xv[i] - m) * rstd;
    dxh[i] = gr[c] * ln_g[c];
    s1 += dxh[i];
    s2 += dxh[i] * xh[i];
  }
  s1 = waveReduceSum(s1) * (1.f / 128.f);
  s2 = waveReduceSum(s2) * (1.f / 128.f);
  float dX[4], gn2 = 0.f;
#pragma unroll
  for (int i = 0; i < 4; ++i) {
    dX[i] = rstd * (dxh[i] - s1 - xh[i] * s2);
    gn2 += dX[i] * dX[i];
  }
  float gn = sqrtf(waveReduceSum(gn2));
  float c1 = fminf(1.0f / fmaxf(gn, 1e-6f), 1.0f);
  float xn[4], sn2 = 0.f;
#pragma unroll
  for (int i = 0; i < 4; ++i) {
    xn[i] = xv[i] - eta * c1 * dX[i];
    sn2 += xn[i] * xn[i];
  }
  float sn = sqrtf(waveReduceSum(sn2));
  float c2 = fminf(10.0f / fmaxf(sn, 1e-6f), 1.0f);
  float* xo = X + (size_t)wave * 128;
#pragma unroll
  for (int i = 0; i < 4; ++i) xo[lane + 32 * i] = xn[i] * c2;
}

// ================= host side =================
static inline int gemm_blocks(int M, int Nc, int NT) {
  long long ntm = (M + 15) / 16;
  long long ntn = (Nc + 15) / 16;
  long long ntg = (ntn + NT - 1) / NT;
  long long waves = ntm * ntg;
  return (int)((waves + 7) / 8);
}

extern "C" void kernel_launch(void* const* d_in, const int* in_sizes, int n_in,
                              void* d_out, int out_size, void* d_ws, size_t ws_size,
                              hipStream_t stream) {
  const float* x       = (const float*)d_in[0];
  const int*   c2      = (const int*)d_in[1];
  const float* u2      = (const float*)d_in[2];
  const float* eattr   = (const float*)d_in[3];
  const int*   c3      = (const int*)d_in[4];
  const int*   v3      = (const int*)d_in[5];
  const float* u3      = (const float*)d_in[6];
  const int*   ttau    = (const int*)d_in[7];
  const float* enc_W1  = (const float*)d_in[8];
  const float* enc_b1  = (const float*)d_in[9];
  const float* enc_g1  = (const float*)d_in[10];
  const float* enc_be1 = (const float*)d_in[11];
  const float* enc_W2  = (const float*)d_in[12];
  const float* enc_b2  = (const float*)d_in[13];
  const float* enc_g2  = (const float*)d_in[14];
  const float* enc_be2 = (const float*)d_in[15];
  const float* em_W1   = (const float*)d_in[16];
  const float* em_b1   = (const float*)d_in[17];
  const float* em_W2   = (const float*)d_in[18];
  const float* em_b2   = (const float*)d_in[19];
  const float* ln_g    = (const float*)d_in[20];
  const float* ln_b    = (const float*)d_in[21];
  const float* W_Q2    = (const float*)d_in[22];
  const float* W_K2    = (const float*)d_in[23];
  const float* W_Q3    = (const float*)d_in[24];
  const float* W_K3    = (const float*)d_in[25];
  const float* T_tau   = (const float*)d_in[26];
  const float* W_Qm    = (const float*)d_in[27];
  const float* W_Km    = (const float*)d_in[28];
  const float* B_mem   = (const float*)d_in[29];
  const float* l2p     = (const float*)d_in[30];
  const float* l3p     = (const float*)d_in[31];
  const float* lmp     = (const float*)d_in[32];
  const float* b2p     = (const float*)d_in[33];
  const float* b3p     = (const float*)d_in[34];
  const float* bmp     = (const float*)d_in[35];
  const float* eta_l   = (const float*)d_in[36];
  const float* ro_W1   = (const float*)d_in[37];
  const float* ro_b1   = (const float*)d_in[38];
  const float* ro_g    = (const float*)d_in[39];
  const float* ro_be   = (const float*)d_in[40];
  const float* ro_W2   = (const float*)d_in[41];
  const float* ro_b2   = (const float*)d_in[42];

  const int N   = in_sizes[0] / 64;
  const int E2n = in_sizes[2];
  const int E3n = in_sizes[6];

  // ---- workspace carve (floats) ----
  float* ws = (float*)d_ws;
  size_t off = 0;
  float* X      = ws + off; off += (size_t)N * 128;
  float* G      = ws + off; off += (size_t)N * 128;
  float* meanv  = ws + off; off += N;
  float* rstdv  = ws + off; off += N;
  float* QK     = ws + off; off += (size_t)N * 640;  // [Q2|K2|Q3|K3|Qm]; encoder scratch too
  float* dQK    = ws + off; off += (size_t)N * 640;
  float* dGb    = ws + off; off += (size_t)N * 128;  // dE/dG accum; readout hidden too
  float* a2     = ws + off; off += E2n;
  float* Km     = ws + off; off += 32 * 128;
  float* Wf     = ws + off; off += 128 * 640;        // Bt for backward GEMM
  float* Wb     = ws + off; off += 640 * 128;        // Bt for forward GEMM
  float* eW1t   = ws + off; off += 256 * 64;
  float* eW2t   = ws + off; off += 128 * 256;
  float* rW1t   = ws + off; off += 128 * 128;
  float* rW2t   = ws + off; off += 10 * 128;
  (void)ws_size;

  const int TB = 256;
  int rowBlocks = (N * 32 + TB - 1) / TB;  // one wave per row
  int e2Blocks  = (E2n + 7) / 8;
  int e3Blocks  = (E3n + 7) / 8;

  // ---- setup: packed / transposed weights ----
  pack_weights_kernel<<<(5 * 128 * 128 + TB - 1) / TB, TB, 0, stream>>>(
      W_Q2, W_K2, W_Q3, W_K3, W_Qm, Wf, Wb);
  transpose_kernel<<<(64 * 256 + TB - 1) / TB, TB, 0, stream>>>(enc_W1, eW1t, 64, 256);
  transpose_kernel<<<(256 * 128 + TB - 1) / TB, TB, 0, stream>>>(enc_W2, eW2t, 256, 128);
  transpose_kernel<<<(128 * 128 + TB - 1) / TB, TB, 0, stream>>>(ro_W1, rW1t, 128, 128);
  transpose_kernel<<<(128 * 10 + TB - 1) / TB, TB, 0, stream>>>(ro_W2, rW2t, 128, 10);
  km_kernel<<<32, 128, 0, stream>>>(B_mem, W_Km, Km);
  edge_bias_kernel<<<e2Blocks, TB, 0, stream>>>(eattr, em_W1, em_b1, em_W2, em_b2, a2, E2n);

  // ---- node encoder: X0 = LN(GELU(LN(GELU(x@W1+b1))@W2+b2)) ----
  gemm_wmma_kernel<2, 1, true, false><<<gemm_blocks(N, 256, 2), TB, 0, stream>>>(
      x, eW1t, enc_b1, QK, N, 256, 64);
  ln_rows_kernel<<<rowBlocks, TB, 0, stream>>>(QK, QK, enc_g1, enc_be1, N, 256, nullptr, nullptr);
  gemm_wmma_kernel<2, 1, true, false><<<gemm_blocks(N, 128, 2), TB, 0, stream>>>(
      QK, eW2t, enc_b2, X, N, 128, 256);
  ln_rows_kernel<<<rowBlocks, TB, 0, stream>>>(X, X, enc_g2, enc_be2, N, 128, nullptr, nullptr);

  // ---- 4 energy descent steps ----
  for (int step = 0; step < 4; ++step) {
    ln_rows_kernel<<<rowBlocks, TB, 0, stream>>>(X, G, ln_g, ln_b, N, 128, meanv, rstdv);
    // fused projections: QK = G @ [WQ2|WK2|WQ3|WK3|WQm]   (N x 128 x 640)
    gemm_wmma_kernel<2, 0, false, false><<<gemm_blocks(N, 640, 2), TB, 0, stream>>>(
        G, Wb, nullptr, QK, N, 640, 128);
    hipMemsetAsync(dQK, 0, (size_t)N * 640 * sizeof(float), stream);
    hipMemsetAsync(dGb, 0, (size_t)N * 128 * sizeof(float), stream);
    edge2_kernel<<<e2Blocks, TB, 0, stream>>>(c2, u2, a2, QK, dQK, l2p, b2p, E2n);
    edge3_kernel<<<e3Blocks, TB, 0, stream>>>(c3, v3, u3, ttau, T_tau, G, QK, dQK, dGb,
                                              l3p, b3p, E3n);
    mem_kernel<<<rowBlocks, TB, 0, stream>>>(QK, Km, dQK, lmp, bmp, N);
    // fused backward projections: dG += dQK @ W^T stacks  (N x 640 x 128)
    gemm_wmma_kernel<2, 0, false, true><<<gemm_blocks(N, 128, 2), TB, 0, stream>>>(
        dQK, Wf, nullptr, dGb, N, 128, 640);
    grad_step_kernel<<<rowBlocks, TB, 0, stream>>>(X, dGb, meanv, rstdv, ln_g, eta_l, N);
  }

  // ---- readout: out = LN2(GELU(LN(X)@roW1+b1)) @ roW2 + b2 ----
  ln_rows_kernel<<<rowBlocks, TB, 0, stream>>>(X, G, ln_g, ln_b, N, 128, nullptr, nullptr);
  gemm_wmma_kernel<2, 1, true, false><<<gemm_blocks(N, 128, 2), TB, 0, stream>>>(
      G, rW1t, ro_b1, dGb, N, 128, 128);
  ln_rows_kernel<<<rowBlocks, TB, 0, stream>>>(dGb, dGb, ro_g, ro_be, N, 128, nullptr, nullptr);
  gemm_wmma_kernel<1, 0, true, false><<<gemm_blocks(N, 10, 1), TB, 0, stream>>>(
      dGb, rW2t, ro_b2, (float*)d_out, N, 10, 128);
  (void)n_in; (void)out_size;
}
